// Attention_7670811590880
// MI455X (gfx1250) — matrix operand order repaired and checked by reading, not compiled, not run on hardware
//
#include <hip/hip_runtime.h>
#include <hip/hip_bf16.h>
#include <cstdint>

#define DEV_INLINE __device__ __forceinline__

typedef __attribute__((ext_vector_type(8)))  float        v8f;
typedef __attribute__((ext_vector_type(16))) __bf16       v16bf;
typedef __attribute__((ext_vector_type(4)))  unsigned int v4u;
typedef __attribute__((ext_vector_type(8)))  int          v8i;
typedef __attribute__((ext_vector_type(4)))  int          v4i;

constexpr int S   = 2048;
constexpr int D   = 4096;
constexpr int H   = 32;
constexpr int KVH = 8;
constexpr int HD  = 128;

// ---------------------------------------------------------------- helpers

DEV_INLINE unsigned short f2bfu(float f) {
  unsigned u = __float_as_uint(f);
  return (unsigned short)((u + 0x7FFFu + ((u >> 16) & 1u)) >> 16);
}
DEV_INLINE __bf16 f2bf(float f) {
  unsigned short h = f2bfu(f);
  return __builtin_bit_cast(__bf16, h);
}

union FragBf {
  v16bf v;
  uint4 u4[2];
};

// 16-bit A-matrix 16x32 fragment: per lane, 16 contiguous
// K = khalf*16 + {0..15} (matches B fragment order).
DEV_INLINE v16bf load_a_frag(const __bf16* p) {
  FragBf f;
  f.u4[0] = *reinterpret_cast<const uint4*>(p);
  f.u4[1] = *reinterpret_cast<const uint4*>(p + 8);
  return f.v;
}
// B fragment from K-contiguous (transposed) staging: lane holds 16 contiguous K.
DEV_INLINE v16bf load_b_frag(const __bf16* p) {
  FragBf f;
  f.u4[0] = *reinterpret_cast<const uint4*>(p);
  f.u4[1] = *reinterpret_cast<const uint4*>(p + 8);
  return f.v;
}

DEV_INLINE v8f wmma_bf16(v16bf a, v16bf b, v8f c) {
  return __builtin_amdgcn_wmma_f32_16x16x32_bf16(
      false, a, false, b, (short)0, c, false, false);
}

DEV_INLINE v8f vzero8() {
  v8f z = {0.f, 0.f, 0.f, 0.f, 0.f, 0.f, 0.f, 0.f};
  return z;
}

// ---------------------------------------------------------------- TDM
// 2-D tile DMA: global (row stride = stride0 elems, tile_d0 contiguous bf16
// per row, tile_d1 rows) -> LDS at lds_off, with optional LDS row padding.
// D# layout per CDNA5 ISA 8.3/8.4. Issued by one wave; EXEC is ignored.
// This toolchain's builtin takes 6 args (4 SGPR groups + spare group + cpol).
DEV_INLINE void tdm_load_2d(unsigned lds_off, const __bf16* gptr,
                            unsigned tile_d0, unsigned tile_d1,
                            unsigned long long stride0,
                            unsigned pad_interval_code, unsigned pad_amount_code,
                            bool pad_en) {
  unsigned long long ga = (unsigned long long)(uintptr_t)gptr;
  v4u g0;
  g0[0] = 1u;                                              // count=1, user mode
  g0[1] = lds_off;                                         // lds_addr
  g0[2] = (unsigned)ga;                                    // global_addr[31:0]
  g0[3] = (unsigned)((ga >> 32) & 0x01FFFFFFu) | (2u << 30); // addr[56:32], type=2
  v8i g1;
  unsigned dw0 = (1u << 16);                               // data_size = 2 bytes
  if (pad_en)
    dw0 |= (1u << 20) | (pad_interval_code << 22) | (pad_amount_code << 25);
  g1[0] = (int)dw0;
  g1[1] = (int)((tile_d0 & 0xFFFFu) << 16);                // tensor_dim0[15:0]
  g1[2] = (int)((tile_d0 >> 16) | ((tile_d1 & 0xFFFFu) << 16)); // dim0 hi | dim1 lo
  g1[3] = (int)((tile_d1 >> 16) | (tile_d0 << 16));        // dim1 hi | tile_dim0
  g1[4] = (int)(tile_d1 & 0xFFFFu);                        // tile_dim1 (tile_dim2=0)
  g1[5] = (int)(stride0 & 0xFFFFFFFFu);                    // tensor_dim0_stride lo
  g1[6] = (int)((stride0 >> 32) & 0xFFFFu);                // stride hi (dim1_stride=0)
  g1[7] = 0;
  v4i g2 = {0, 0, 0, 0};
  v4i g3 = {0, 0, 0, 0};
  v8i g4 = {0, 0, 0, 0, 0, 0, 0, 0};
  __builtin_amdgcn_tensor_load_to_lds(g0, g1, g2, g3, g4, 0);
}

// ---------------------------------------------------------------- convert

__global__ __launch_bounds__(256)
void f32_to_bf16_x4(const float4* __restrict__ in, ushort* __restrict__ out, size_t n4) {
  size_t i = (size_t)blockIdx.x * blockDim.x + threadIdx.x;
  if (i < n4) {
    float4 f = in[i];
    ushort4 o4;
    o4.x = f2bfu(f.x); o4.y = f2bfu(f.y); o4.z = f2bfu(f.z); o4.w = f2bfu(f.w);
    *reinterpret_cast<ushort4*>(out + 4 * i) = o4;
  }
}

// ---------------------------------------------------------------- RoPE (+scale, +bf16)

__global__ __launch_bounds__(256)
void rope_to_bf16(const float* __restrict__ in, const float* __restrict__ cosT,
                  const float* __restrict__ sinT, __bf16* __restrict__ out,
                  int nheads, float scale) {
  int i = blockIdx.x * blockDim.x + threadIdx.x;
  int total = S * nheads * (HD / 2);
  if (i >= total) return;
  int fi = i % (HD / 2);
  int h  = (i / (HD / 2)) % nheads;
  int s  = i / ((HD / 2) * nheads);
  size_t base = ((size_t)s * nheads + h) * HD + 2 * fi;
  float te = in[base], to = in[base + 1];
  float c  = cosT[s * (HD / 2) + fi];
  float sn = sinT[s * (HD / 2) + fi];
  out[base]     = f2bf((te * c - to * sn) * scale);
  out[base + 1] = f2bf((to * c + te * sn) * scale);
}

// ---------------------------------------------------------------- WMMA GEMM
// C[M,N](f32) = A[M,K](bf16 row-major) x B[K,N](bf16 row-major).
// N,K compile-time so all strides fold into immediate offsets.
// Block 128x128, BK=32, 8 waves (2x4), wave tile 64x32 (4x2 WMMA tiles).
// Ping-pong LDS: the TDM moves A tile k+1 and threads stage B tile k+1
// (transposed) while tile k feeds the WMMAs; one barrier per K-step.

constexpr int BM = 128, BN = 128, BK = 32;

template <int N, int K>
__global__ __launch_bounds__(256)
void gemm_bf16_f32(const __bf16* __restrict__ A, const __bf16* __restrict__ B,
                   float* __restrict__ C, int M) {
  __shared__ __bf16 As[2][BM][BK + 8];   // row-major, K contiguous (TDM pads +8)
  __shared__ __bf16 Bt[2][BN][BK + 8];   // transposed: N-major, K contiguous

  const int tid  = threadIdx.x;
  const int lane = tid & 31;
  const int wid  = tid >> 5;
  const int r    = lane & 15;
  const int kh   = lane >> 4;
  const int wm   = (wid >> 2) * 64;
  const int wn   = (wid & 3) * 32;
  const int bm   = blockIdx.y * BM;
  const int bn   = blockIdx.x * BN;

  // A tile DMA: 128 rows x 32 bf16, row stride K elems.
  // LDS pad: row = 16 DWORDs (code 3), pad = 4 DWORDs (code 3).
  auto stageA = [&](int k0, int buf) {
    tdm_load_2d((unsigned)(uintptr_t)&As[buf][0][0], A + (size_t)bm * K + k0,
                BK, BM, (unsigned long long)K, 3u, 3u, true);
  };
  // B tile transposed by hand (TDM cannot transpose).
  auto stageB = [&](int k0, int buf) {
#pragma unroll
    for (int i = 0; i < 2; ++i) {
      int chunk = tid + i * 256;
      int krow = chunk >> 4;
      int ncol = (chunk & 15) * 8;
      const __bf16* src = B + (size_t)(k0 + krow) * N + bn + ncol;
      uint4 d = *reinterpret_cast<const uint4*>(src);
      if (k0 + 2 * BK < K) __builtin_prefetch(src + (size_t)BK * N, 0, 1);
      const __bf16* dsm = reinterpret_cast<const __bf16*>(&d);
#pragma unroll
      for (int j = 0; j < 8; ++j) Bt[buf][ncol + j][krow] = dsm[j];
    }
  };

  v8f acc[4][2];
#pragma unroll
  for (int mt = 0; mt < 4; ++mt)
#pragma unroll
    for (int nt = 0; nt < 2; ++nt) acc[mt][nt] = vzero8();

  // Prologue: stage tile 0.
  if (wid == 0) stageA(0, 0);
  stageB(0, 0);
  if (wid == 0) __builtin_amdgcn_s_wait_tensorcnt(0);
  __syncthreads();

  constexpr int NK = K / BK;
  for (int kt = 0; kt < NK; ++kt) {
    const int cur = kt & 1, nxt = cur ^ 1;

    // Kick off tile kt+1 into the other buffer before computing tile kt.
    if (kt + 1 < NK) {
      if (wid == 0) stageA((kt + 1) * BK, nxt);
      stageB((kt + 1) * BK, nxt);
    }

    // Compute tile kt.
    v16bf bfrag[2];
#pragma unroll
    for (int nt = 0; nt < 2; ++nt)
      bfrag[nt] = load_b_frag(&Bt[cur][wn + nt * 16 + r][kh * 16]);
#pragma unroll
    for (int mt = 0; mt < 4; ++mt) {
      v16bf afrag = load_a_frag(&As[cur][wm + mt * 16 + r][kh * 16]);
#pragma unroll
      for (int nt = 0; nt < 2; ++nt)
        acc[mt][nt] = wmma_bf16(afrag, bfrag[nt], acc[mt][nt]);
    }

    if (kt + 1 < NK && wid == 0) __builtin_amdgcn_s_wait_tensorcnt(0);
    __syncthreads();
  }

  // Store with compile-time row stride -> immediate offsets.
#pragma unroll
  for (int mt = 0; mt < 4; ++mt)
#pragma unroll
    for (int nt = 0; nt < 2; ++nt) {
      int col = bn + wn + nt * 16 + r;
      int rowbase = bm + wm + mt * 16 + ((lane < 16) ? 0 : 8);
      float* p = C + (size_t)rowbase * N + col;
#pragma unroll
      for (int e = 0; e < 8; ++e) p[(size_t)e * N] = acc[mt][nt][e];
    }
}

// ---------------------------------------------------------------- flash attention
// Grid: (S/64, H). Block 128 = 4 waves; wave w owns queries q0..q0+15.
// K tile staged by TDM; V staged transposed by hand; Q pre-scaled bf16.

constexpr int QB = 64, KB = 64;

__global__ __launch_bounds__(128)
void flash_attn(const __bf16* __restrict__ q,   // [S][H][HD]
                const __bf16* __restrict__ k,   // [S][KVH][HD]
                const __bf16* __restrict__ v,   // [S][KVH][HD]
                __bf16* __restrict__ o) {       // [S][H*HD]
  __shared__ __bf16 Ks[KB][HD + 8];        // keys row-major (TDM pads +8)
  __shared__ __bf16 Vt[HD][KB + 8];        // V transposed
  __shared__ __bf16 Ps[4][16][KB + 8];     // per-wave P scratch

  const int tid  = threadIdx.x;
  const int lane = tid & 31;
  const int wid  = tid >> 5;
  const int r    = lane & 15;
  const int kh   = lane >> 4;
  const int h    = blockIdx.y;
  const int qb   = blockIdx.x;
  const int hk   = h >> 2;
  const int q0   = qb * QB + wid * 16;

  v16bf qf[4];
#pragma unroll
  for (int t = 0; t < 4; ++t)
    qf[t] = load_a_frag(q + ((size_t)(q0 + r) * H + h) * HD + t * 32 + kh * 16);

  v8f Oacc[8];
#pragma unroll
  for (int ot = 0; ot < 8; ++ot) Oacc[ot] = vzero8();
  float m[8], l[8];
#pragma unroll
  for (int e = 0; e < 8; ++e) { m[e] = -3.0e38f; l[e] = 0.f; }

  for (int kb = 0; kb <= qb; ++kb) {
    // K tile via TDM: 64 rows x 128 bf16, row stride KVH*HD.
    // LDS pad: row = 64 DWORDs (code 5), pad = 4 DWORDs (code 3).
    if (wid == 0)
      tdm_load_2d((unsigned)(uintptr_t)&Ks[0][0],
                  k + ((size_t)(kb * KB) * KVH + hk) * HD,
                  HD, KB, (unsigned long long)(KVH * HD), 5u, 3u, true);

    // V transposed staging: [feature][key].
    for (int c = tid; c < (KB * HD) / 8; c += 128) {
      int key = c >> 4;
      int f0  = (c & 15) * 8;
      uint4 d = *reinterpret_cast<const uint4*>(
          v + ((size_t)(kb * KB + key) * KVH + hk) * HD + f0);
      const __bf16* ds = reinterpret_cast<const __bf16*>(&d);
#pragma unroll
      for (int j = 0; j < 8; ++j) Vt[f0 + j][key] = ds[j];
    }
    if (wid == 0) __builtin_amdgcn_s_wait_tensorcnt(0);
    __syncthreads();

    // Scores: 4 key tiles of 16, 4 WMMA K-steps over HD=128.
    v8f sc[4];
#pragma unroll
    for (int nt = 0; nt < 4; ++nt) {
      sc[nt] = vzero8();
#pragma unroll
      for (int t = 0; t < 4; ++t) {
        v16bf bk = load_b_frag(&Ks[nt * 16 + r][t * 32 + kh * 16]);
        sc[nt] = wmma_bf16(qf[t], bk, sc[nt]);
      }
    }

    if (kb == qb) {  // causal mask, diagonal block only
#pragma unroll
      for (int nt = 0; nt < 4; ++nt) {
        int kcol = kb * KB + nt * 16 + r;
#pragma unroll
        for (int e = 0; e < 8; ++e) {
          int qrow = q0 + ((lane < 16) ? e : 8 + e);
          if (kcol > qrow) sc[nt][e] += -1.0e9f;
        }
      }
    }

    // Online softmax per row (row e lives across 16 lanes of a half-wave).
#pragma unroll
    for (int e = 0; e < 8; ++e) {
      float mt_ = sc[0][e];
#pragma unroll
      for (int nt = 1; nt < 4; ++nt) mt_ = fmaxf(mt_, sc[nt][e]);
      for (int off = 1; off < 16; off <<= 1)
        mt_ = fmaxf(mt_, __shfl_xor(mt_, off, 32));
      float mnew = fmaxf(m[e], mt_);
      float corr = __expf(m[e] - mnew);
      float rs = 0.f;
#pragma unroll
      for (int nt = 0; nt < 4; ++nt) {
        float p_ = __expf(sc[nt][e] - mnew);
        sc[nt][e] = p_;
        rs += p_;
      }
      for (int off = 1; off < 16; off <<= 1) rs += __shfl_xor(rs, off, 32);
      l[e] = l[e] * corr + rs;
      m[e] = mnew;
#pragma unroll
      for (int ot = 0; ot < 8; ++ot) Oacc[ot][e] *= corr;
    }

    // P (C-layout) -> per-wave LDS -> A-fragments for PV.
#pragma unroll
    for (int nt = 0; nt < 4; ++nt)
#pragma unroll
      for (int e = 0; e < 8; ++e) {
        int rowM = (lane < 16) ? e : 8 + e;
        Ps[wid][rowM][nt * 16 + r] = f2bf(sc[nt][e]);
      }
    asm volatile("s_wait_dscnt 0" ::: "memory");

#pragma unroll
    for (int pt = 0; pt < 2; ++pt) {
      v16bf pf = load_a_frag(&Ps[wid][r][pt * 32 + kh * 16]);
#pragma unroll
      for (int ot = 0; ot < 8; ++ot) {
        v16bf bv = load_b_frag(&Vt[ot * 16 + r][pt * 32 + kh * 16]);
        Oacc[ot] = wmma_bf16(pf, bv, Oacc[ot]);
      }
    }
    __syncthreads();
  }

  // Epilogue with immediate-offset stores (stride H*HD is compile-time).
#pragma unroll
  for (int ot = 0; ot < 8; ++ot) {
    int col = h * HD + ot * 16 + r;
    __bf16* p = o + (size_t)(q0 + ((lane < 16) ? 0 : 8)) * (H * HD) + col;
#pragma unroll
    for (int e = 0; e < 8; ++e) p[(size_t)e * (H * HD)] = f2bf(Oacc[ot][e] / l[e]);
  }
}

// ---------------------------------------------------------------- launch

extern "C" void kernel_launch(void* const* d_in, const int* in_sizes, int n_in,
                              void* d_out, int out_size, void* d_ws, size_t ws_size,
                              hipStream_t stream) {
  (void)in_sizes; (void)n_in; (void)out_size; (void)ws_size;
  const float* x    = (const float*)d_in[0];
  const float* wq   = (const float*)d_in[1];
  const float* wk   = (const float*)d_in[2];
  const float* wv   = (const float*)d_in[3];
  const float* wo   = (const float*)d_in[4];
  const float* fcos = (const float*)d_in[5];
  const float* fsin = (const float*)d_in[6];
  float* out = (float*)d_out;

  char* ws = (char*)d_ws;
  size_t off = 0;
  auto carve = [&](size_t elems, size_t esz) -> void* {
    void* p = ws + off;
    off += ((elems * esz) + 255) & ~(size_t)255;
    return p;
  };
  __bf16* x_bf  = (__bf16*)carve((size_t)S * D, 2);
  __bf16* wq_bf = (__bf16*)carve((size_t)D * H * HD, 2);
  __bf16* wk_bf = (__bf16*)carve((size_t)D * KVH * HD, 2);
  __bf16* wv_bf = (__bf16*)carve((size_t)D * KVH * HD, 2);
  __bf16* wo_bf = (__bf16*)carve((size_t)H * HD * D, 2);
  float*  qf32  = (float*) carve((size_t)S * H * HD, 4);
  float*  kf32  = (float*) carve((size_t)S * KVH * HD, 4);
  float*  vf32  = (float*) carve((size_t)S * KVH * HD, 4);
  __bf16* q_bf  = (__bf16*)carve((size_t)S * H * HD, 2);
  __bf16* k_bf  = (__bf16*)carve((size_t)S * KVH * HD, 2);
  __bf16* v_bf  = (__bf16*)carve((size_t)S * KVH * HD, 2);
  __bf16* a_bf  = (__bf16*)carve((size_t)S * H * HD, 2);

  auto cvt = [&](const float* src, __bf16* dst, size_t n) {
    size_t n4 = n / 4;
    int blocks = (int)((n4 + 255) / 256);
    f32_to_bf16_x4<<<dim3(blocks), dim3(256), 0, stream>>>(
        (const float4*)src, (ushort*)dst, n4);
  };
  cvt(x,  x_bf,  (size_t)S * D);
  cvt(wq, wq_bf, (size_t)D * H * HD);
  cvt(wk, wk_bf, (size_t)D * KVH * HD);
  cvt(wv, wv_bf, (size_t)D * KVH * HD);
  cvt(wo, wo_bf, (size_t)H * HD * D);

  // QKV projections (bf16 WMMA, f32 out). K dim is always D=4096.
  gemm_bf16_f32<H * HD, D><<<dim3((H * HD) / BN, S / BM), 256, 0, stream>>>(x_bf, wq_bf, qf32, S);
  gemm_bf16_f32<KVH * HD, D><<<dim3((KVH * HD) / BN, S / BM), 256, 0, stream>>>(x_bf, wk_bf, kf32, S);
  gemm_bf16_f32<KVH * HD, D><<<dim3((KVH * HD) / BN, S / BM), 256, 0, stream>>>(x_bf, wv_bf, vf32, S);

  // RoPE + bf16 (Q carries 1/sqrt(HD)); V straight convert.
  {
    int totq = S * H * (HD / 2);
    int totk = S * KVH * (HD / 2);
    rope_to_bf16<<<dim3((totq + 255) / 256), dim3(256), 0, stream>>>(
        qf32, fcos, fsin, q_bf, H, 0.08838834764831845f);
    rope_to_bf16<<<dim3((totk + 255) / 256), dim3(256), 0, stream>>>(
        kf32, fcos, fsin, k_bf, KVH, 1.0f);
    cvt(vf32, v_bf, (size_t)S * KVH * HD);
  }

  // Causal GQA flash attention.
  flash_attn<<<dim3(S / QB, H), 128, 0, stream>>>(q_bf, k_bf, v_bf, a_bf);

  // Output projection -> f32 result (K dim = H*HD = 4096).
  gemm_bf16_f32<D, H * HD><<<dim3(D / BN, S / BM), 256, 0, stream>>>(a_bf, wo_bf, out, S);
}
